// LocalAttentionLayer_11321533792397
// MI455X (gfx1250) — matrix operand order repaired and checked by reading, not compiled, not run on hardware
//
#include <hip/hip_runtime.h>
#include <math.h>

typedef __attribute__((ext_vector_type(16))) _Float16 v16h;
typedef __attribute__((ext_vector_type(8)))  float    v8f;
typedef __attribute__((ext_vector_type(4)))  unsigned int u32x4;
typedef __attribute__((ext_vector_type(8)))  int i32x8;
typedef __attribute__((ext_vector_type(4)))  int i32x4;

using half_t = _Float16;

#define HWIN 6
#define BSZ  2
#define SEQ  4096
#define CH   768
#define NH   12
#define HD   64
#define CQKV 2304   // 3*CH

// LDS A-tile geometry: 32 rows of 1536B data + 48B pad (3 pads of 16B per row,
// inserted by TDM every 512B of streamed data) -> row stride 1584B.
#define LDS_A_ROWB   1584
#define LDS_A_HALVES (32 * (LDS_A_ROWB / 2))

__device__ __forceinline__ int iclamp(int v, int lo, int hi) {
  return v < lo ? lo : (v > hi ? hi : v);
}

// Load a 16-half fragment as two 16B chunks (generic pointer: global or LDS).
__device__ __forceinline__ v16h load_frag(const half_t* p0, const half_t* p1) {
  v16h r;
  uint4* rp = reinterpret_cast<uint4*>(&r);
  rp[0] = *reinterpret_cast<const uint4*>(p0);
  rp[1] = *reinterpret_cast<const uint4*>(p1);
  return r;
}

// Address of half element (row, kh) inside the TDM-padded LDS A tile.
// inrow bytes = kh*2; a 16B pad is inserted after every 512B of data.
__device__ __forceinline__ const half_t* lds_a(const half_t* sA, int row, int kh) {
  const int inrow = kh * 2;
  const int off = row * LDS_A_ROWB + inrow + ((inrow >> 9) << 4);
  return reinterpret_cast<const half_t*>(reinterpret_cast<const char*>(sA) + off);
}

// ---- TDM: DMA a 32 x 768 f16 tile (rows contiguous, row pitch 1536B) into LDS.
// D# per CDNA5 ISA ch.8: data_size=8B units, tensor_dim0=192, tile=192x32,
// stride=192, pad_enable, pad_interval=512B (enc 6), pad_amount=16B (enc 3).
__device__ __forceinline__ void tdm_load_tile_a(const half_t* gsrc, unsigned lds_off) {
  unsigned long long ga = (unsigned long long)(uintptr_t)gsrc;
  u32x4 g0 = {};
  g0[0] = 1u;                                        // count=1, user mode
  g0[1] = lds_off;                                   // lds_addr (bytes)
  g0[2] = (unsigned)(ga & 0xFFFFFFFFull);            // global_addr[31:0]
  g0[3] = (unsigned)((ga >> 32) & 0x01FFFFFFull)     // global_addr[56:32]
          | (2u << 30);                              // type=2 ("image")
  i32x8 g1 = {};
  g1[0] = (int)((3u << 16)      // data_size: 8B units
              | (1u << 20)      // pad_enable
              | (6u << 22)      // pad_interval: 128 DW = 512B
              | (3u << 25));    // pad_amount: 4 DW = 16B
  g1[1] = (int)(192u << 16);    // tensor_dim0[15:0] = 192 (bits 63:48)
  g1[2] = (int)(32u  << 16);    // tensor_dim1[15:0] = 32  (bits 95:80)
  g1[3] = (int)(192u << 16);    // tile_dim0 = 192        (bits 127:112)
  g1[4] = 32;                   // tile_dim1 = 32          (bits 143:128)
  g1[5] = 192;                  // tensor_dim0_stride[31:0] = 192 units
  i32x4 g2 = {}, g3 = {};
#if __clang_major__ >= 23
  i32x8 g4 = {};
  __builtin_amdgcn_tensor_load_to_lds(g0, g1, g2, g3, g4, 0);
#else
  __builtin_amdgcn_tensor_load_to_lds(g0, g1, g2, g3, 0);
#endif
}

// ---------------- cast / transpose helpers ----------------

__global__ void k_cast_f32_to_f16(const float* __restrict__ in,
                                  half_t* __restrict__ out, int n) {
  int i = blockIdx.x * blockDim.x + threadIdx.x;
  if (i < n) out[i] = (half_t)in[i];
}

// in: [K][Nn] f32 row-major  ->  out: [Nn][K] f16 (transposed)
__global__ void k_transpose_cast(const float* __restrict__ in,
                                 half_t* __restrict__ out, int K, int Nn) {
  int i = blockIdx.x * blockDim.x + threadIdx.x;
  if (i < K * Nn) {
    int nn = i / K;
    int k  = i % K;
    out[i] = (half_t)in[(size_t)k * Nn + nn];
  }
}

// ---------------- wave-level GEMM core (A tile in LDS) ----------------
// sA: TDM-staged 32xK f16 tile.  BT: [N][K] f16 (B transposed) in global.
// Wave computes 32(M) x 64(N) with fp32 accumulation.
__device__ __forceinline__ void wave_gemm_32x64_ldsA(const half_t* __restrict__ sA,
                                                     const half_t* __restrict__ BT,
                                                     int K, int n0, v8f acc[2][4]) {
  const int l    = threadIdx.x & 31;
  const int rsel = l & 15;
  const int akb  = (l < 16) ? 0 : 8;    // A-frag K sub-offset
  const int bkb  = (l < 16) ? 0 : 16;   // unused: B frags use akb, as A does

  for (int k0 = 0; k0 < K; k0 += 32) {
    v16h a[2];
#pragma unroll
    for (int i = 0; i < 2; ++i) {
      const int row = i * 16 + rsel;
      a[i] = load_frag(lds_a(sA, row, k0 + akb), lds_a(sA, row, k0 + 16 + akb));
    }
    v16h b[4];
#pragma unroll
    for (int j = 0; j < 4; ++j) {
      const half_t* base = BT + (size_t)(n0 + j * 16 + rsel) * K + k0 + akb;
      b[j] = load_frag(base, base + 16);
    }
#pragma unroll
    for (int i = 0; i < 2; ++i)
#pragma unroll
      for (int j = 0; j < 4; ++j)
        acc[i][j] = __builtin_amdgcn_wmma_f32_16x16x32_f16(
            false, a[i], false, b[j], (short)0, acc[i][j], false, false);
  }
}

// ---------------- QKV GEMM: qkv = x @ w_qkv, scatter to Q/K/Vt ----------------
__global__ __launch_bounds__(128) void k_qkv_gemm(
    const half_t* __restrict__ X, const half_t* __restrict__ WT,
    half_t* __restrict__ Q, half_t* __restrict__ Kbuf,
    half_t* __restrict__ Vt) {
  __shared__ alignas(16) half_t sA[LDS_A_HALVES];
  const int wave = threadIdx.x >> 5;
  const int l    = threadIdx.x & 31;
  const int m0   = blockIdx.x * 32;
  const int n0   = blockIdx.y * 256 + wave * 64;

  if (wave == 0) {
    tdm_load_tile_a(X + (size_t)m0 * CH, (unsigned)(uintptr_t)sA);
    __builtin_amdgcn_s_wait_tensorcnt(0);
  }
  __syncthreads();

  v8f acc[2][4] = {};
  wave_gemm_32x64_ldsA(sA, WT, CH, n0, acc);

  const int col0  = l & 15;
  const int rbase = (l >> 4) << 3;   // 0 or 8
#pragma unroll
  for (int i = 0; i < 2; ++i) {
#pragma unroll
    for (int j = 0; j < 4; ++j) {
      const int c     = n0 + j * 16 + col0;
      const int which = c / CH;        // 0=q 1=k 2=v
      const int rem   = c % CH;
      const int h     = rem >> 6;
      const int d     = rem & 63;
#pragma unroll
      for (int e = 0; e < 8; ++e) {
        const int m  = m0 + i * 16 + rbase + e;
        const int b  = m >> 12;        // /SEQ
        const int n  = m & (SEQ - 1);
        const size_t bh = (size_t)(b * NH + h);
        const half_t v  = (half_t)acc[i][j][e];
        if (which == 0)      Q   [(bh * SEQ + n) * HD + d] = v;
        else if (which == 1) Kbuf[(bh * SEQ + n) * HD + d] = v;
        else                 Vt  [(bh * HD + d) * SEQ + n] = v;
      }
    }
  }
}

// ---------------- windowed attention: one wave per (b,h, 16-query block) ----------------
__global__ __launch_bounds__(32) void k_local_attn(
    const half_t* __restrict__ Q, const half_t* __restrict__ Kbuf,
    const half_t* __restrict__ Vt, half_t* __restrict__ Attn) {
  __shared__ float  sS[16][32];
  __shared__ half_t sP[16][32];

  const int bh    = blockIdx.x >> 8;   // SEQ/16 = 256 blocks per (b,h)
  const int qb    = blockIdx.x & 255;
  const int n0    = qb * 16;
  const int kbase = n0 - HWIN;         // first key of the 28-wide window

  const int l    = threadIdx.x & 31;
  const int rsel = l & 15;
  const int akb  = (l < 16) ? 0 : 8;
  const int bkb  = (l < 16) ? 0 : 16;
  const int rb   = (l >> 4) << 3;

  const half_t* Qp = Q    + (size_t)bh * SEQ * HD;
  const half_t* Kp = Kbuf + (size_t)bh * SEQ * HD;
  const half_t* Vp = Vt   + (size_t)bh * HD * SEQ;

  // ---- scores: S(16x32) = Q(16xHD) @ K_win^T, two 16-key tiles, HD split 2x32
  v8f sc[2] = {};
#pragma unroll
  for (int ks = 0; ks < 2; ++ks) {
    const half_t* qrow = Qp + (size_t)(n0 + rsel) * HD + ks * 32;
    v16h af = load_frag(qrow + akb, qrow + 16 + akb);
#pragma unroll
    for (int t = 0; t < 2; ++t) {
      int key = iclamp(kbase + t * 16 + rsel, 0, SEQ - 1);  // clamped; masked in softmax
      const half_t* krow = Kp + (size_t)key * HD + ks * 32 + akb;
      v16h bf = load_frag(krow, krow + 16);
      sc[t] = __builtin_amdgcn_wmma_f32_16x16x32_f16(
          false, af, false, bf, (short)0, sc[t], false, false);
    }
  }
#pragma unroll
  for (int t = 0; t < 2; ++t)
#pragma unroll
    for (int e = 0; e < 8; ++e)
      sS[rb + e][t * 16 + rsel] = sc[t][e];
  __syncthreads();

  // ---- masked softmax: lanes 0..15, one query row each; valid jj in [row, row+12]
  if (l < 16) {
    const int row = l;
    float v[28];
    float mx = -3.0e38f;
#pragma unroll
    for (int jj = 0; jj < 28; ++jj) {
      const int key = kbase + jj;
      const bool valid = (jj >= row) && (jj <= row + 2 * HWIN) &&
                         (key >= 0) && (key < SEQ);
      const float s = valid ? sS[row][jj] * 0.125f : -3.0e38f;  // scale = HD^-0.5
      v[jj] = s;
      mx = fmaxf(mx, s);
    }
    float sum = 0.f;
#pragma unroll
    for (int jj = 0; jj < 28; ++jj) {
      const float e = __expf(v[jj] - mx);
      v[jj] = e;
      sum += e;
    }
    const float inv = 1.0f / sum;
#pragma unroll
    for (int jj = 0; jj < 28; ++jj) sP[row][jj] = (half_t)(v[jj] * inv);
#pragma unroll
    for (int jj = 28; jj < 32; ++jj) sP[row][jj] = (half_t)0.f;
  }
  __syncthreads();

  // ---- out = P(16x32) @ V_win(32xHD): single K-step, 4 d-tiles of 16
  const half_t* prow = &sP[rsel][0];
  v16h pf = load_frag(prow + bkb, prow + bkb + 8);

  const bool interior = (kbase >= 0) && (kbase + 32 <= SEQ);
  const int  b = bh / NH, h = bh % NH;
#pragma unroll
  for (int j = 0; j < 4; ++j) {
    const int d = j * 16 + rsel;
    const half_t* vrow = Vp + (size_t)d * SEQ;
    v16h vf;
    if (interior) {
      // window base only even-aligned -> dword loads
      uint* vp = reinterpret_cast<uint*>(&vf);
      const uint* gp = reinterpret_cast<const uint*>(vrow + kbase + bkb);
#pragma unroll
      for (int q = 0; q < 8; ++q) vp[q] = gp[q];
    } else {
      half_t* hv = reinterpret_cast<half_t*>(&vf);
#pragma unroll
      for (int q = 0; q < 16; ++q) {
        const int key = iclamp(kbase + bkb + q, 0, SEQ - 1);  // P==0 on invalid cols
        hv[q] = vrow[key];
      }
    }
    v8f o = {};
    o = __builtin_amdgcn_wmma_f32_16x16x32_f16(
        false, pf, false, vf, (short)0, o, false, false);
#pragma unroll
    for (int e = 0; e < 8; ++e) {
      const int n = n0 + rb + e;
      Attn[((size_t)(b * SEQ + n)) * CH + h * HD + j * 16 + rsel] = (half_t)o[e];
    }
  }
}

// ---------------- proj GEMM: out = attn @ w_proj + b_proj (fp32 out) ----------------
__global__ __launch_bounds__(128) void k_proj_gemm(
    const half_t* __restrict__ A, const half_t* __restrict__ WT,
    const float* __restrict__ bias, float* __restrict__ Out) {
  __shared__ alignas(16) half_t sA[LDS_A_HALVES];
  const int wave = threadIdx.x >> 5;
  const int l    = threadIdx.x & 31;
  const int m0   = blockIdx.x * 32;
  const int n0   = blockIdx.y * 256 + wave * 64;

  if (wave == 0) {
    tdm_load_tile_a(A + (size_t)m0 * CH, (unsigned)(uintptr_t)sA);
    __builtin_amdgcn_s_wait_tensorcnt(0);
  }
  __syncthreads();

  v8f acc[2][4] = {};
  wave_gemm_32x64_ldsA(sA, WT, CH, n0, acc);

  const int col0  = l & 15;
  const int rbase = (l >> 4) << 3;
#pragma unroll
  for (int i = 0; i < 2; ++i) {
#pragma unroll
    for (int j = 0; j < 4; ++j) {
      const int   c  = n0 + j * 16 + col0;
      const float bv = bias[c];
#pragma unroll
      for (int e = 0; e < 8; ++e) {
        const int m = m0 + i * 16 + rbase + e;
        Out[(size_t)m * CH + c] = acc[i][j][e] + bv;
      }
    }
  }
}

// ---------------- host launcher ----------------
extern "C" void kernel_launch(void* const* d_in, const int* in_sizes, int n_in,
                              void* d_out, int out_size, void* d_ws, size_t ws_size,
                              hipStream_t stream) {
  (void)in_sizes; (void)n_in; (void)out_size; (void)ws_size;
  const float* x      = (const float*)d_in[0];
  const float* w_qkv  = (const float*)d_in[1];
  const float* w_proj = (const float*)d_in[2];
  const float* b_proj = (const float*)d_in[3];
  float* out = (float*)d_out;

  char* ws = (char*)d_ws;
  size_t off = 0;
  auto take = [&](size_t bytes) -> void* {
    void* p = ws + off;
    off += (bytes + 255) & ~(size_t)255;
    return p;
  };
  const size_t tokC = (size_t)BSZ * SEQ * CH;        // 6.29M elems
  half_t* Xh     = (half_t*)take(tokC * 2);
  half_t* WqkvT  = (half_t*)take((size_t)CQKV * CH * 2);
  half_t* WprojT = (half_t*)take((size_t)CH * CH * 2);
  half_t* Qb     = (half_t*)take(tokC * 2);
  half_t* Kb     = (half_t*)take(tokC * 2);
  half_t* Vt     = (half_t*)take(tokC * 2);
  half_t* Attn   = (half_t*)take(tokC * 2);

  const int nx = (int)tokC;
  k_cast_f32_to_f16<<<(nx + 255) / 256, 256, 0, stream>>>(x, Xh, nx);
  k_transpose_cast<<<(CQKV * CH + 255) / 256, 256, 0, stream>>>(w_qkv, WqkvT, CH, CQKV);
  k_transpose_cast<<<(CH * CH + 255) / 256, 256, 0, stream>>>(w_proj, WprojT, CH, CH);

  k_qkv_gemm<<<dim3(BSZ * SEQ / 32, CQKV / 256), 128, 0, stream>>>(Xh, WqkvT, Qb, Kb, Vt);
  k_local_attn<<<dim3(BSZ * NH * (SEQ / 16)), 32, 0, stream>>>(Qb, Kb, Vt, Attn);
  k_proj_gemm<<<dim3(BSZ * SEQ / 32, CH / 256), 128, 0, stream>>>(Attn, WprojT, b_proj, out);
}
